// LSTM_Transformer_86234353369622
// MI455X (gfx1250) — matrix operand order repaired and checked by reading, not compiled, not run on hardware
//
#include <hip/hip_runtime.h>

// ---------------------------------------------------------------------------
// CDNA5 (gfx1250) types / helpers
// ---------------------------------------------------------------------------
typedef __attribute__((ext_vector_type(16))) __bf16          bf16x16;
typedef __attribute__((ext_vector_type(8)))  float           floatx8;
typedef __attribute__((ext_vector_type(8)))  unsigned short  ushort8;

#define DEV_INLINE __device__ __forceinline__

DEV_INLINE unsigned short f2bf(float x) {            // round-to-nearest-even
    unsigned u = __float_as_uint(x);
    u += 0x7fffu + ((u >> 16) & 1u);
    return (unsigned short)(u >> 16);
}

// Async DMA: global -> LDS, 16B per lane, tracked by ASYNCcnt (CDNA5 path).
// lds_off is a byte offset inside the dynamic-LDS block (base 0, no static LDS).
DEV_INLINE void async_copy_b128(unsigned lds_off, const void* gaddr) {
    asm volatile("global_load_async_to_lds_b128 %0, %1, off"
                 :: "v"(lds_off), "v"(gaddr) : "memory");
}
DEV_INLINE void wait_async_le2() { asm volatile("s_wait_asynccnt 0x2" ::: "memory"); }
DEV_INLINE void wait_async_0()   { asm volatile("s_wait_asynccnt 0x0" ::: "memory"); }

// B-matrix 32x16 bf16 fragment from a row-major [N,K] weight array (global).
DEV_INLINE bf16x16 load_b_frag(const unsigned short* W, int ldw, int n0, int k0, int lane) {
    int col = n0 + (lane & 15);
    int kb  = k0 + ((lane >> 4) << 4);
    const unsigned short* p = W + (size_t)col * ldw + kb;
    union { bf16x16 v; ushort8 h[2]; } u;
    u.h[0] = *(const ushort8*)(p);
    u.h[1] = *(const ushort8*)(p + 8);
    return u.v;
}
DEV_INLINE floatx8 wmma_bf16(bf16x16 a, bf16x16 b, floatx8 c) {
    return __builtin_amdgcn_wmma_f32_16x16x32_bf16(false, a, false, b, (short)0, c, false, false);
}

// ---------------------------------------------------------------------------
// fp32 -> bf16 conversion
// ---------------------------------------------------------------------------
__global__ __launch_bounds__(256) void k_f32_to_bf16(const float* __restrict__ x,
                                                     unsigned short* __restrict__ y, size_t n) {
    size_t i = (size_t)blockIdx.x * 256 + threadIdx.x;
    if (i < n) y[i] = f2bf(x[i]);
}

// ---------------------------------------------------------------------------
// Generic batched bf16 WMMA GEMM:  C[bh] = act(A[bh] @ W[bh]^T + bias + bias2 + resid)
// W is [N,K] row-major (torch Linear layout).
// Block = 256 thr (8 waves) -> 64(M) x 32(N) tile, wave = one 16x16 WMMA tile.
// Double-buffered async-LDS pipeline: per 32-K stage the block DMAs
//   A tile 64x32 bf16 (4KB) + B tile 32x32 bf16 (2KB) into LDS with
//   global_load_async_to_lds_b128; each wave issues exactly 2 async ops/stage,
//   so "s_wait_asynccnt 2" keeps the newest stage in flight while retiring the
//   older one -> DMA overlaps WMMA.  12KB LDS/block keeps WGP occupancy high.
// ---------------------------------------------------------------------------
__global__ __launch_bounds__(256) void k_gemm(
    const unsigned short* __restrict__ A, int lda,
    const unsigned short* __restrict__ W, int ldw,
    const float* __restrict__ bias, const float* __restrict__ bias2,
    const float* __restrict__ resid, int ldr,
    float* __restrict__ C, unsigned short* __restrict__ Cbf, int ldc,
    int M, int N, int K, int act, int nh,
    size_t aB, size_t wB, size_t cB, int aH, int wH, int cH)
{
    extern __shared__ char lds[];   // [2][6144]: A tile 4KB + B tile 2KB per buffer

    int bh = blockIdx.z, b = bh / nh, h = bh - b * nh;
    A += (size_t)b * aB + (size_t)h * aH;
    W += (size_t)b * wB + (size_t)h * wH;
    size_t coff = (size_t)b * cB + (size_t)h * cH;

    const int tid  = threadIdx.x;
    const int wave = tid >> 5, lane = tid & 31;
    const int m0base = blockIdx.y * 64;
    const int n0base = blockIdx.x * 32;
    const int wm16 = (wave & 3) * 16;        // wave's M sub-tile inside As
    const int wn16 = (wave >> 2) * 16;       // wave's N sub-tile inside Bs

    auto stage = [&](int kk, int buf) {
        // A tile: 64 rows x 32 halves (each thread 16B)
        {
            int row = tid >> 2, kq = (tid & 3) << 3;
            unsigned loff = (unsigned)(buf * 6144 + row * 64 + kq * 2);
            async_copy_b128(loff, A + (size_t)(m0base + row) * lda + kk + kq);
        }
        // B tile: 32 cols x 32 halves (lanes 0..15 of every wave -> 1 op/wave)
        if (lane < 16) {
            int chunk = wave * 16 + lane;    // 0..127
            int col = chunk >> 2, kq = (chunk & 3) << 3;
            unsigned loff = (unsigned)(buf * 6144 + 4096 + col * 64 + kq * 2);
            async_copy_b128(loff, W + (size_t)(n0base + col) * ldw + kk + kq);
        }
        // long-range prefetch of the weight stream into GL2
        if (kk + 32 < K)
            __builtin_prefetch(W + (size_t)(n0base + (tid & 31)) * ldw + kk + 32, 0, 1);
    };

    floatx8 acc = {};
    int buf = 0;
    stage(0, 0);
    for (int k0 = 0; k0 < K; k0 += 32) {
        __syncthreads();                       // all waves done reading buf^1
        bool more = (k0 + 32 < K);
        if (more) stage(k0 + 32, buf ^ 1);     // DMA next stage while computing
        if (more) wait_async_le2();            // retire current stage's 2 ops
        else      wait_async_0();
        __syncthreads();                       // stage data visible block-wide

        const unsigned short* As = (const unsigned short*)(lds + buf * 6144);
        const unsigned short* Bs = (const unsigned short*)(lds + buf * 6144 + 4096);

        bf16x16 a, bm;
        {   // A-frag (ISA 7.12.2 16-bit A 16x32): lane&15 = M row
            int row = lane & 15, kb = (lane >> 4) << 4;
            const unsigned short* p = As + (wm16 + row) * 32 + kb;
            union { bf16x16 v; ushort8 hh[2]; } ua;
            ua.hh[0] = *(const ushort8*)(p);
            ua.hh[1] = *(const ushort8*)(p + 8);
            a = ua.v;
        }
        {   // B-frag: lane&15 = N col, lanes>=16 hold K+16
            int col = lane & 15, kb = (lane >> 4) << 4;
            const unsigned short* p = Bs + (wn16 + col) * 32 + kb;
            union { bf16x16 v; ushort8 hh[2]; } ub;
            ub.hh[0] = *(const ushort8*)(p);
            ub.hh[1] = *(const ushort8*)(p + 8);
            bm = ub.v;
        }
        acc = wmma_bf16(a, bm, acc);
        buf ^= 1;
    }

    int n0 = n0base + wn16, m0 = m0base + wm16;
    int n = n0 + (lane & 15);
    float bv = 0.f;
    if (bias)  bv += bias[n];
    if (bias2) bv += bias2[n];
    int mb = m0 + ((lane >> 4) << 3);
    for (int r = 0; r < 8; ++r) {
        int m = mb + r;
        float v = acc[r] + bv;
        if (resid) v += resid[(size_t)m * ldr + n];
        if (act)   v = fmaxf(v, 0.f);
        size_t idx = coff + (size_t)m * ldc + n;
        C[idx] = v;
        if (Cbf) Cbf[idx] = f2bf(v);
    }
}

// ---------------------------------------------------------------------------
// Causal softmax over rows of [nbh, S, S]:  p = softmax(sc*scale + (k>q ? -1e9 : 0))
// writes bf16 probabilities for the P@V GEMM.
// ---------------------------------------------------------------------------
__global__ __launch_bounds__(256) void k_softmax_causal(const float* __restrict__ Sc,
                                                        unsigned short* __restrict__ Pbf,
                                                        int S, float scale)
{
    int q = blockIdx.x;
    size_t base = ((size_t)blockIdx.y * S + q) * (size_t)S;
    __shared__ float red[8];
    int tid = threadIdx.x;

    float mx = -3.0e38f;
    for (int k = tid; k < S; k += 256) {
        float v = Sc[base + k] * scale + (k > q ? -1e9f : 0.0f);
        mx = fmaxf(mx, v);
    }
    for (int o = 16; o; o >>= 1) mx = fmaxf(mx, __shfl_xor(mx, o, 32));
    if ((tid & 31) == 0) red[tid >> 5] = mx;
    __syncthreads();
    float rmax = red[tid & 7];
    for (int o = 4; o; o >>= 1) rmax = fmaxf(rmax, __shfl_xor(rmax, o, 32));
    __syncthreads();

    float sum = 0.f;
    for (int k = tid; k < S; k += 256) {
        float v = Sc[base + k] * scale + (k > q ? -1e9f : 0.0f);
        sum += __expf(v - rmax);
    }
    for (int o = 16; o; o >>= 1) sum += __shfl_xor(sum, o, 32);
    if ((tid & 31) == 0) red[tid >> 5] = sum;
    __syncthreads();
    float tot = red[tid & 7];
    for (int o = 4; o; o >>= 1) tot += __shfl_xor(tot, o, 32);
    float inv = 1.0f / tot;

    for (int k = tid; k < S; k += 256) {
        float v = Sc[base + k] * scale + (k > q ? -1e9f : 0.0f);
        Pbf[base + k] = f2bf(__expf(v - rmax) * inv);
    }
}

// ---------------------------------------------------------------------------
// LayerNorm rows of length H; fp32 out + optional bf16 shadow
// ---------------------------------------------------------------------------
__global__ __launch_bounds__(256) void k_layernorm(const float* __restrict__ X,
                                                   const float* __restrict__ g,
                                                   const float* __restrict__ bta,
                                                   float* __restrict__ Y,
                                                   unsigned short* __restrict__ Ybf, int H)
{
    size_t base = (size_t)blockIdx.x * H;
    __shared__ float rs[8], rs2[8];
    int tid = threadIdx.x;
    float s = 0.f, s2 = 0.f;
    for (int i = tid; i < H; i += 256) { float v = X[base + i]; s += v; s2 += v * v; }
    for (int o = 16; o; o >>= 1) { s += __shfl_xor(s, o, 32); s2 += __shfl_xor(s2, o, 32); }
    if ((tid & 31) == 0) { rs[tid >> 5] = s; rs2[tid >> 5] = s2; }
    __syncthreads();
    float ts = rs[tid & 7], ts2 = rs2[tid & 7];
    for (int o = 4; o; o >>= 1) { ts += __shfl_xor(ts, o, 32); ts2 += __shfl_xor(ts2, o, 32); }
    float mean = ts / H;
    float var  = ts2 / H - mean * mean;
    float inv  = rsqrtf(var + 1e-5f);
    for (int i = tid; i < H; i += 256) {
        float v = (X[base + i] - mean) * inv * g[i] + bta[i];
        Y[base + i] = v;
        if (Ybf) Ybf[base + i] = f2bf(v);
    }
}

// ---------------------------------------------------------------------------
// Build Vt[bh][d][k] = bf16(in[b][k][h*inHead + d])  (so P@V B-frags stay contiguous)
// ---------------------------------------------------------------------------
__global__ void k_transpose_bf(const float* __restrict__ in, unsigned short* __restrict__ out,
                               int nh, int S, int D, int inRow, int inHead, long long inBatch)
{
    int bh = blockIdx.z, b = bh / nh, h = bh - b * nh;
    int k = blockIdx.x * 16 + threadIdx.x;
    int d = blockIdx.y * 16 + threadIdx.y;
    if (k < S && d < D) {
        float v = in[(size_t)b * (size_t)inBatch + (size_t)k * inRow + h * inHead + d];
        out[((size_t)bh * D + d) * S + k] = f2bf(v);
    }
}

// ---------------------------------------------------------------------------
// Persistent single-WGP LSTM layer.  B=8, H=1024, 4H=4096, 32 waves.
// h kept in LDS as bf16 16x1024 tile (rows 8..15 zero); gates staged in LDS.
// Whh (8MB bf16) streamed each step -> resident in 192MB L2 across 512 steps.
// Dynamic LDS: 8*4096*4 + 16*1024*2 + 2*8*1024*4 = 224KB (<320KB WGP LDS).
// ---------------------------------------------------------------------------
__global__ __launch_bounds__(1024) void k_lstm_layer(
    const float* __restrict__ xg, const unsigned short* __restrict__ Whh,
    const float* __restrict__ h0, const float* __restrict__ c0,
    float* __restrict__ y, unsigned short* __restrict__ ybf,
    float* __restrict__ hN, float* __restrict__ cN, int S)
{
    extern __shared__ char smem[];
    float*          g_lds = (float*)smem;                                   // [8][4096]
    unsigned short* h_bf  = (unsigned short*)(smem + 8 * 4096 * 4);         // [16][1024]
    float*          h_f   = (float*)(smem + 8 * 4096 * 4 + 16 * 1024 * 2);  // [8][1024]
    float*          c_f   = h_f + 8 * 1024;                                 // [8][1024]

    const int tid = threadIdx.x, lane = tid & 31, wave = tid >> 5;

    for (int i = tid; i < 16 * 1024; i += 1024) h_bf[i] = 0;
    __syncthreads();
    for (int i = tid; i < 8 * 1024; i += 1024) {
        float hv = h0[i];
        h_f[i] = hv; c_f[i] = c0[i]; h_bf[i] = f2bf(hv);
    }
    __syncthreads();

    for (int t = 0; t < S; ++t) {
        // gates[8,4096] = h @ Whh^T  (+ xg on store)
        for (int j = 0; j < 8; ++j) {
            int n0 = (wave * 8 + j) * 16;
            floatx8 acc = {};
            for (int k0 = 0; k0 < 1024; k0 += 32) {
                bf16x16 a;
                {
                    int row = lane & 15, kb = k0 + ((lane >> 4) << 4);
                    const unsigned short* p = h_bf + row * 1024 + kb;
                    union { bf16x16 v; ushort8 hh[2]; } ua;
                    ua.hh[0] = *(const ushort8*)(p);
                    ua.hh[1] = *(const ushort8*)(p + 8);
                    a = ua.v;
                }
                bf16x16 bm = load_b_frag(Whh, 1024, n0, k0, lane);
                acc = wmma_bf16(a, bm, acc);
            }
            int n = n0 + (lane & 15);
            int mb = (lane >> 4) << 3;
            for (int r = 0; r < 8; ++r) {
                int m = mb + r;
                if (m < 8)
                    g_lds[m * 4096 + n] = acc[r] + xg[((size_t)m * S + t) * 4096 + n];
            }
        }
        __syncthreads();
        // fused cell update; gate order i,f,g,o
        {
            int jh = tid;
            for (int b = 0; b < 8; ++b) {
                float ig = g_lds[b * 4096 + jh];
                float fg = g_lds[b * 4096 + 1024 + jh];
                float gg = g_lds[b * 4096 + 2048 + jh];
                float og = g_lds[b * 4096 + 3072 + jh];
                ig = 1.f / (1.f + __expf(-ig));
                fg = 1.f / (1.f + __expf(-fg));
                og = 1.f / (1.f + __expf(-og));
                gg = tanhf(gg);
                float c  = fg * c_f[b * 1024 + jh] + ig * gg;
                float hh = og * tanhf(c);
                c_f[b * 1024 + jh] = c;
                h_f[b * 1024 + jh] = hh;
                h_bf[b * 1024 + jh] = f2bf(hh);
                size_t oi = ((size_t)b * S + t) * 1024 + jh;
                y[oi] = hh;
                if (ybf) ybf[oi] = f2bf(hh);
            }
        }
        __syncthreads();
    }
    if (hN) {
        for (int i = tid; i < 8 * 1024; i += 1024) { hN[i] = h_f[i]; cN[i] = c_f[i]; }
    }
}

// ---------------------------------------------------------------------------
// fused = (wt*t1 + wl*lstm + wa*attn + wt2*t2) / (wl+wa+wt+wt2)  -> bf16
// ---------------------------------------------------------------------------
__global__ void k_fuse(const float* __restrict__ t1, const float* __restrict__ t2,
                       const float* __restrict__ at, const float* __restrict__ ls,
                       const float* __restrict__ wl, const float* __restrict__ wt,
                       const float* __restrict__ wt2, const float* __restrict__ wa,
                       unsigned short* __restrict__ ob, size_t n)
{
    float a = wl[0], bq = wt[0], c = wt2[0], d = wa[0];
    float inv = 1.f / (a + d + bq + c);
    for (size_t i = (size_t)blockIdx.x * blockDim.x + threadIdx.x; i < n;
         i += (size_t)gridDim.x * blockDim.x)
        ob[i] = f2bf((bq * t1[i] + a * ls[i] + d * at[i] + c * t2[i]) * inv);
}

// ---------------------------------------------------------------------------
// log_softmax over rows of 128
// ---------------------------------------------------------------------------
__global__ __launch_bounds__(128) void k_logsoftmax(const float* __restrict__ X,
                                                    float* __restrict__ Y, int N)
{
    int r = blockIdx.x, tid = threadIdx.x;
    float v = X[(size_t)r * N + tid];
    __shared__ float red[4];
    float mx = v;
    for (int o = 16; o; o >>= 1) mx = fmaxf(mx, __shfl_xor(mx, o, 32));
    if ((tid & 31) == 0) red[tid >> 5] = mx;
    __syncthreads();
    float m2 = red[tid & 3];
    for (int o = 2; o; o >>= 1) m2 = fmaxf(m2, __shfl_xor(m2, o, 32));
    __syncthreads();
    float e = __expf(v - m2), sum = e;
    for (int o = 16; o; o >>= 1) sum += __shfl_xor(sum, o, 32);
    if ((tid & 31) == 0) red[tid >> 5] = sum;
    __syncthreads();
    float s2 = red[tid & 3];
    for (int o = 2; o; o >>= 1) s2 += __shfl_xor(s2, o, 32);
    Y[(size_t)r * N + tid] = v - m2 - __logf(s2);
}

// ---------------------------------------------------------------------------
// Host orchestration
// ---------------------------------------------------------------------------
extern "C" void kernel_launch(void* const* d_in, const int* in_sizes, int n_in,
                              void* d_out, int out_size, void* d_ws, size_t ws_size,
                              hipStream_t stream)
{
    (void)n_in; (void)out_size; (void)ws_size;
    const int B = 8, S = 512, IN = 512, H = 1024, NH = 16, HD = 64, G = 4096, M = 4096;

    size_t off = 0;
    auto alloc = [&](size_t bytes) -> void* {
        void* p = (char*)d_ws + off;
        off = (off + bytes + 255) & ~(size_t)255;
        return p;
    };
    auto fp = [&](int i) { return (const float*)d_in[i]; };
    auto cvt = [&](int idx) -> const unsigned short* {
        size_t n = (size_t)in_sizes[idx];
        unsigned short* p = (unsigned short*)alloc(n * 2);
        k_f32_to_bf16<<<dim3((unsigned)((n + 255) / 256)), dim3(256), 0, stream>>>(fp(idx), p, n);
        return p;
    };

    // ---- weights -> bf16 (input dict flat order) ----
    const unsigned short* fcw1 = cvt(3);
    const unsigned short* fcw2 = cvt(5);
    const unsigned short *enc_inw[2][2], *enc_outw[2][2], *enc_ffw1[2][2], *enc_ffw2[2][2];
    int encBase[2] = {7, 31};
    for (int e = 0; e < 2; ++e)
        for (int l = 0; l < 2; ++l) {
            int base = encBase[e] + l * 12;
            enc_inw[e][l]  = cvt(base + 0);
            enc_outw[e][l] = cvt(base + 2);
            enc_ffw1[e][l] = cvt(base + 6);
            enc_ffw2[e][l] = cvt(base + 8);
        }
    const unsigned short* l1_wih[2] = {cvt(55), cvt(59)};
    const unsigned short* l1_whh[2] = {cvt(56), cvt(60)};
    const unsigned short* l2_wih[2] = {cvt(63), cvt(67)};
    const unsigned short* l2_whh[2] = {cvt(64), cvt(68)};
    const unsigned short* fc1w1 = cvt(71);
    const unsigned short* fc1w2 = cvt(73);
    const unsigned short* ow1   = cvt(75);
    const unsigned short* ow2   = cvt(77);
    const unsigned short* in_bf = cvt(0);

    // ---- activation scratch ----
    float* x1  = (float*)alloc((size_t)M * 1024 * 4);  unsigned short* x1b  = (unsigned short*)alloc((size_t)M * 1024 * 2);
    float* li  = (float*)alloc((size_t)M * H * 4);     unsigned short* lib  = (unsigned short*)alloc((size_t)M * H * 2);
    float* qkv = (float*)alloc((size_t)M * 3 * H * 4); unsigned short* qkvb = (unsigned short*)alloc((size_t)M * 3 * H * 2);
    float* sc  = (float*)alloc((size_t)B * NH * S * S * 4);
    unsigned short* pbf = (unsigned short*)alloc((size_t)B * NH * S * S * 2);
    unsigned short* vtb = (unsigned short*)alloc((size_t)B * H * S * 2); // covers both 128x64x512 and 8x1024x512
    float* o   = (float*)alloc((size_t)M * H * 4);     unsigned short* ob2  = (unsigned short*)alloc((size_t)M * H * 2);
    float* tmp = (float*)alloc((size_t)M * H * 4);
    float* ff1 = (float*)alloc((size_t)M * 4 * H * 4); unsigned short* ff1b = (unsigned short*)alloc((size_t)M * 4 * H * 2);
    float* t1  = (float*)alloc((size_t)M * H * 4);     unsigned short* t1b  = (unsigned short*)alloc((size_t)M * H * 2);
    float* t2  = (float*)alloc((size_t)M * H * 4);     unsigned short* t2b  = (unsigned short*)alloc((size_t)M * H * 2);
    float* attnO = (float*)alloc((size_t)M * H * 4);
    float* xg  = (float*)alloc((size_t)M * G * 4);
    float* y1  = (float*)alloc((size_t)M * H * 4);     unsigned short* y1b  = (unsigned short*)alloc((size_t)M * H * 2);
    float* lstmO = (float*)alloc((size_t)M * H * 4);
    unsigned short* fusedb = (unsigned short*)alloc((size_t)M * H * 2);
    float* yA = (float*)alloc((size_t)M * 512 * 4);    unsigned short* yAb = (unsigned short*)alloc((size_t)M * 512 * 2);
    float* yB = (float*)alloc((size_t)M * 512 * 4);    unsigned short* yBb = (unsigned short*)alloc((size_t)M * 512 * 2);
    float* z1 = (float*)alloc((size_t)M * H * 4);      unsigned short* z1b = (unsigned short*)alloc((size_t)M * H * 2);
    float* fin = (float*)alloc((size_t)M * H * 4);     unsigned short* finb = (unsigned short*)alloc((size_t)M * H * 2);
    float* u  = (float*)alloc((size_t)M * 512 * 4);    unsigned short* ub  = (unsigned short*)alloc((size_t)M * 512 * 2);
    float* logits = (float*)alloc((size_t)M * 128 * 4);

    const size_t gemm_smem = 2 * 6144;   // double-buffered A+B tiles

    auto gemm = [&](const unsigned short* A, int lda, const unsigned short* W, int ldw,
                    const float* bias, const float* bias2, const float* resid, int ldr,
                    float* C, unsigned short* Cbf, int ldc, int Mm, int Nn, int Kk, int act,
                    int nbh = 1, int nh = 1, size_t aB = 0, size_t wB = 0, size_t cB = 0,
                    int aH = 0, int wH = 0, int cH = 0) {
        dim3 g((Nn + 31) / 32, (Mm + 63) / 64, nbh);
        k_gemm<<<g, dim3(256), gemm_smem, stream>>>(A, lda, W, ldw, bias, bias2, resid, ldr,
                                                    C, Cbf, ldc, Mm, Nn, Kk, act, nh,
                                                    aB, wB, cB, aH, wH, cH);
    };

    // ---- input FC: relu(relu(x@w1^T+b1)@w2^T+b2) ----
    gemm(in_bf, IN, fcw1, IN, fp(4), nullptr, nullptr, 0, x1, x1b, 1024, M, 1024, IN, 1);
    gemm(x1b, 1024, fcw2, 1024, fp(6), nullptr, nullptr, 0, li, lib, H, M, H, 1024, 1);

    // ---- custom single-head causal attention (unscaled) ----
    gemm(lib, H, lib, H, nullptr, nullptr, nullptr, 0, sc, nullptr, S, S, S, H, 0,
         B, 1, (size_t)S * H, (size_t)S * H, (size_t)S * S, 0, 0, 0);
    k_softmax_causal<<<dim3(S, B), dim3(256), 0, stream>>>(sc, pbf, S, 1.0f);
    k_transpose_bf<<<dim3(S / 16, H / 16, B), dim3(16, 16), 0, stream>>>(li, vtb, 1, S, H, H, 0, (long long)S * H);
    gemm(pbf, S, vtb, S, nullptr, nullptr, nullptr, 0, attnO, nullptr, H, S, H, S, 0,
         B, 1, (size_t)S * S, (size_t)H * S, (size_t)S * H, 0, 0, 0);

    // ---- two post-norm transformer encoders (2 layers each) ----
    for (int e = 0; e < 2; ++e) {
        float* xo = e ? t2 : t1;
        unsigned short* xob = e ? t2b : t1b;
        for (int l = 0; l < 2; ++l) {
            int base = encBase[e] + l * 12;
            const float* xin = l ? xo : li;
            const unsigned short* xinb = l ? (const unsigned short*)xob : lib;
            // QKV
            gemm(xinb, H, enc_inw[e][l], H, fp(base + 1), nullptr, nullptr, 0,
                 qkv, qkvb, 3 * H, M, 3 * H, H, 0);
            // scores = Q K^T per (b,h)
            gemm(qkvb, 3 * H, qkvb + H, 3 * H, nullptr, nullptr, nullptr, 0, sc, nullptr, S,
                 S, S, HD, 0, B * NH, NH, (size_t)S * 3 * H, (size_t)S * 3 * H,
                 (size_t)NH * S * S, HD, HD, S * S);
            k_softmax_causal<<<dim3(S, B * NH), dim3(256), 0, stream>>>(sc, pbf, S, 0.125f);
            k_transpose_bf<<<dim3(S / 16, HD / 16, B * NH), dim3(16, 16), 0, stream>>>(
                qkv + 2 * H, vtb, NH, S, HD, 3 * H, HD, (long long)S * 3 * H);
            // O = P V per (b,h)
            gemm(pbf, S, vtb, S, nullptr, nullptr, nullptr, 0, o, ob2, H, S, HD, S, 0,
                 B * NH, NH, (size_t)NH * S * S, (size_t)NH * HD * S, (size_t)S * H,
                 S * S, HD * S, HD);
            // out-proj + residual, LN1
            gemm(ob2, H, enc_outw[e][l], H, fp(base + 3), nullptr, xin, H,
                 tmp, nullptr, H, M, H, H, 0);
            k_layernorm<<<dim3(M), dim3(256), 0, stream>>>(tmp, fp(base + 4), fp(base + 5), xo, xob, H);
            // FF + residual, LN2
            gemm(xob, H, enc_ffw1[e][l], H, fp(base + 7), nullptr, nullptr, 0,
                 ff1, ff1b, 4 * H, M, 4 * H, H, 1);
            gemm(ff1b, 4 * H, enc_ffw2[e][l], 4 * H, fp(base + 9), nullptr, xo, H,
                 tmp, nullptr, H, M, H, 4 * H, 0);
            k_layernorm<<<dim3(M), dim3(256), 0, stream>>>(tmp, fp(base + 10), fp(base + 11), xo, xob, H);
        }
    }

    // ---- LSTM stack 1 (states discarded) ----
    const float* gv = fp(1);
    const float* uv = fp(2);
    size_t lstm_smem = (size_t)8 * 4096 * 4 + 16 * 1024 * 2 + 2 * 8 * 1024 * 4; // 224 KB
    gemm(lib, H, l1_wih[0], H, fp(57), fp(58), nullptr, 0, xg, nullptr, G, M, G, H, 0);
    k_lstm_layer<<<dim3(1), dim3(1024), lstm_smem, stream>>>(xg, l1_whh[0], gv, uv, y1, y1b,
                                                             nullptr, nullptr, S);
    gemm(y1b, H, l1_wih[1], H, fp(61), fp(62), nullptr, 0, xg, nullptr, G, M, G, H, 0);
    k_lstm_layer<<<dim3(1), dim3(1024), lstm_smem, stream>>>(xg, l1_whh[1], gv + B * H, uv + B * H,
                                                             lstmO, nullptr, nullptr, nullptr, S);

    // ---- fusion ----
    k_fuse<<<dim3(4096), dim3(256), 0, stream>>>(t1, t2, attnO, lstmO,
                                                 fp(79), fp(80), fp(81), fp(82),
                                                 fusedb, (size_t)M * H);

    // ---- fc1: relu,relu ----
    gemm(fusedb, H, fc1w1, H, fp(72), nullptr, nullptr, 0, yA, yAb, 512, M, 512, H, 1);
    gemm(yAb, 512, fc1w2, 512, fp(74), nullptr, nullptr, 0, yB, yBb, 512, M, 512, 512, 1);

    // ---- LSTM stack 2 (final states -> d_out) ----
    float* outGame = (float*)d_out + (size_t)M * 128;
    float* outUser = outGame + (size_t)2 * B * H;
    gemm(yBb, 512, l2_wih[0], 512, fp(65), fp(66), nullptr, 0, xg, nullptr, G, M, G, 512, 0);
    k_lstm_layer<<<dim3(1), dim3(1024), lstm_smem, stream>>>(xg, l2_whh[0], gv, uv, z1, z1b,
                                                             outGame, outUser, S);
    gemm(z1b, H, l2_wih[1], H, fp(69), fp(70), nullptr, 0, xg, nullptr, G, M, G, H, 0);
    k_lstm_layer<<<dim3(1), dim3(1024), lstm_smem, stream>>>(xg, l2_whh[1], gv + B * H, uv + B * H,
                                                             fin, finb, outGame + B * H, outUser + B * H, S);

    // ---- output head + log_softmax ----
    gemm(finb, H, ow1, H, fp(76), nullptr, nullptr, 0, u, ub, 512, M, 512, H, 1);
    gemm(ub, 512, ow2, 512, fp(78), nullptr, nullptr, 0, logits, nullptr, 128, M, 128, 512, 0);
    k_logsoftmax<<<dim3(M), dim3(128), 0, stream>>>(logits, (float*)d_out, 128);
}
